// LSTMNet_20091857011062
// MI455X (gfx1250) — matrix operand order repaired and checked by reading, not compiled, not run on hardware
//
#include <hip/hip_runtime.h>
#include <hip/hip_bf16.h>

// ---------------------------------------------------------------------------
// Fused 2-layer persistent LSTM for MI455X (gfx1250, wave32, WMMA).
//   B=2048, T=512, H=50 (K padded to 64), gates G=200 (N padded to 208).
//   One wave (32 lanes) owns 16 batch rows through both layers and all 512
//   timesteps. Recurrent matmuls use v_wmma_f32_16x16x32_f16; x staging uses
//   global_load_async_to_lds_b128 (ASYNCcnt) when the toolchain exposes it.
// ---------------------------------------------------------------------------

typedef __attribute__((ext_vector_type(16))) _Float16 v16h;
typedef __attribute__((ext_vector_type(8)))  float    v8f;
typedef int v4i_ __attribute__((vector_size(16)));   // matches builtin param type

#define AS1 __attribute__((address_space(1)))
#define AS3 __attribute__((address_space(3)))

#if __has_builtin(__builtin_amdgcn_global_load_async_to_lds_b128) && \
    __has_builtin(__builtin_amdgcn_s_wait_asynccnt)
#define USE_ASYNC_LDS 1
#endif

#define T_LEN   512
#define HID     50
#define GATES   200
#define KPAD    64
#define NTILE   13      // ceil(208/16)
#define GPAD    212     // padded gate-row stride (bank-conflict avoidance)
#define MROWS   16      // batch rows per wave

// ---- LDS layout (bytes) ----
#define OFF_WB1   0                         // W_hh1 f16 frags: 13*2*32*16 halves
#define SZ_WFRAG  (NTILE * 2 * 32 * 16 * 2) // 26624 B
#define OFF_WI2   (OFF_WB1 + SZ_WFRAG)
#define OFF_WH2   (OFF_WI2 + SZ_WFRAG)
#define OFF_B1    (OFF_WH2 + SZ_WFRAG)      // bias1[208] f32
#define OFF_B2    (OFF_B1 + 208 * 4)
#define OFF_WIH1  (OFF_B2 + 208 * 4)        // W_ih1 column [208] f32
#define OFF_FCW   (OFF_WIH1 + 208 * 4)      // fc_W[64] f32 (pad)
#define OFF_H1    (OFF_FCW + 256)           // h1 A-frags: 2*32*16 halves = 2048 B
#define OFF_H2    (OFF_H1 + 2048)
#define OFF_G     (OFF_H2 + 2048)           // gates scratch [16][212] f32
#define OFF_X     (OFF_G + MROWS * GPAD * 4)// x tile [16][512] f32 = 32768 B
#define SMEM_BYTES (OFF_X + MROWS * T_LEN * 4)  // = 133056

static __device__ __forceinline__ float fsig(float x) {
    // sigmoid(x) = 1 / (1 + exp2(-x * log2(e)))
    return __builtin_amdgcn_rcpf(1.0f + __builtin_amdgcn_exp2f(-1.4426950408889634f * x));
}

static __device__ __forceinline__ float ftanh(float x) {
#if __has_builtin(__builtin_amdgcn_tanhf)
    return __builtin_amdgcn_tanhf(x);       // native v_tanh_f32 on gfx1250
#elif __has_builtin(__builtin_amdgcn_tanh_f32)
    return __builtin_amdgcn_tanh_f32(x);
#else
    float e = __builtin_amdgcn_exp2f(2.885390081777927f * x);  // exp(2x)
    return (e - 1.0f) * __builtin_amdgcn_rcpf(e + 1.0f);
#endif
}

// Fill one f16 B-fragment set from a [200 x 50] f32 weight matrix (row = gate n,
// col = k). Fragment element F -> (tile, khalf, lane, e):
//   column N = tile*16 + (lane & 15)
//   K       = khalf*32 + ((lane >> 4) << 4) + e     (consecutive e = consecutive K)
static __device__ __forceinline__ void fill_bfrag(const float* __restrict__ W,
                                                  _Float16* __restrict__ dst, int lane) {
    for (int F = lane; F < NTILE * 2 * 32 * 16; F += 32) {
        int e  = F & 15;
        int ln = (F >> 4) & 31;
        int kh = (F >> 9) & 1;
        int tl = F >> 10;
        int n  = tl * 16 + (ln & 15);
        int k  = kh * 32 + ((ln >> 4) << 4) + e;
        float v = (n < GATES && k < HID) ? W[n * HID + k] : 0.0f;
        dst[F] = (_Float16)v;
    }
}

// Map hidden index j (0..49) to A-fragment (frag, lane, elem), contiguous K
// layout matching the B fragments: lane m holds K=0..15 (e0..15); lane m+16
// holds K=16..31.
static __device__ __forceinline__ int h_addr(int m, int j) {
    int f  = j >> 5;
    int kk = j & 31;
    int lp = m + ((kk >> 4) << 4);
    int idx = kk & 15;
    return (f * 32 + lp) * 16 + idx;   // half-element index into h buffer
}

__global__ __launch_bounds__(32) void lstm_fused(
    const float* __restrict__ x,
    const float* __restrict__ Wih1, const float* __restrict__ Whh1,
    const float* __restrict__ bih1, const float* __restrict__ bhh1,
    const float* __restrict__ Wih2, const float* __restrict__ Whh2,
    const float* __restrict__ bih2, const float* __restrict__ bhh2,
    const float* __restrict__ fcW,  const float* __restrict__ fcb,
    float* __restrict__ out)
{
    extern __shared__ char smem[];
    _Float16* wB1   = (_Float16*)(smem + OFF_WB1);
    _Float16* wI2   = (_Float16*)(smem + OFF_WI2);
    _Float16* wH2   = (_Float16*)(smem + OFF_WH2);
    float*    bias1 = (float*)(smem + OFF_B1);
    float*    bias2 = (float*)(smem + OFF_B2);
    float*    wih1  = (float*)(smem + OFF_WIH1);
    float*    fcw   = (float*)(smem + OFF_FCW);
    _Float16* hbuf1 = (_Float16*)(smem + OFF_H1);
    _Float16* hbuf2 = (_Float16*)(smem + OFF_H2);
    float*    gates = (float*)(smem + OFF_G);
    float*    xtile = (float*)(smem + OFF_X);

    const int lane = threadIdx.x;
    const int bm   = blockIdx.x * MROWS;   // batch base for this wave

    // ---- startup: stage x tile via async global->LDS DMA (ASYNCcnt path) ----
    {
        const float* xg = x + (size_t)bm * T_LEN;
#ifdef USE_ASYNC_LDS
        for (int i = lane * 4; i < MROWS * T_LEN; i += 32 * 4) {
            __builtin_amdgcn_global_load_async_to_lds_b128(
                (AS1 v4i_*)(void*)(xg + i),
                (AS3 v4i_*)(void*)(xtile + i),
                0, 0);
        }
#else
        for (int i = lane * 4; i < MROWS * T_LEN; i += 32 * 4)
            *(float4*)&xtile[i] = *(const float4*)&xg[i];
#endif
    }

    // Pull weight matrices toward GL2 while we run the conversion loops.
    __builtin_prefetch(Whh1, 0, 0);
    __builtin_prefetch(Wih2, 0, 0);
    __builtin_prefetch(Whh2, 0, 0);

    // ---- convert weights to f16 B-fragments in LDS ----
    fill_bfrag(Whh1, wB1, lane);
    fill_bfrag(Wih2, wI2, lane);
    fill_bfrag(Whh2, wH2, lane);
    for (int n = lane; n < 208; n += 32) {
        bias1[n] = (n < GATES) ? (bih1[n] + bhh1[n]) : 0.0f;
        bias2[n] = (n < GATES) ? (bih2[n] + bhh2[n]) : 0.0f;
        wih1[n]  = (n < GATES) ? Wih1[n] : 0.0f;
    }
    for (int j = lane; j < 64; j += 32) fcw[j] = (j < HID) ? fcW[j] : 0.0f;
    // zero h buffers (incl. K padding region)
    {
        unsigned* z1 = (unsigned*)hbuf1;
        unsigned* z2 = (unsigned*)hbuf2;
        for (int i = lane; i < 512; i += 32) { z1[i] = 0u; z2[i] = 0u; }
    }

#ifdef USE_ASYNC_LDS
    __builtin_amdgcn_s_wait_asynccnt(0);    // x tile resident before first read
#endif

    float c1[25], c2[25];
#pragma unroll
    for (int q = 0; q < 25; ++q) { c1[q] = 0.0f; c2[q] = 0.0f; }

    const int row0 = 8 * (lane >> 4);
    const int colL = lane & 15;

    for (int t = 0; t < T_LEN; ++t) {
        // ================= Layer 1: gates1 = h1 @ W_hh1^T =================
        v16h a0 = *(const v16h*)(hbuf1 + lane * 16);
        v16h a1 = *(const v16h*)(hbuf1 + (32 + lane) * 16);
#pragma unroll
        for (int tl = 0; tl < NTILE; ++tl) {
            v16h b0 = *(const v16h*)(wB1 + ((tl * 2 + 0) * 32 + lane) * 16);
            v16h b1 = *(const v16h*)(wB1 + ((tl * 2 + 1) * 32 + lane) * 16);
            v8f acc = {0.f, 0.f, 0.f, 0.f, 0.f, 0.f, 0.f, 0.f};
            acc = __builtin_amdgcn_wmma_f32_16x16x32_f16(false, a0, false, b0,
                                                         (short)0, acc, false, false);
            acc = __builtin_amdgcn_wmma_f32_16x16x32_f16(false, a1, false, b1,
                                                         (short)0, acc, false, false);
            int col = tl * 16 + colL;
#pragma unroll
            for (int v = 0; v < 8; ++v) gates[(row0 + v) * GPAD + col] = acc[v];
        }
        // ---- layer-1 elementwise: add x*W_ih1 + bias, gate nonlinearities ----
#pragma unroll
        for (int q = 0; q < 25; ++q) {
            int id = lane + 32 * q;            // 0..799 = (m, j)
            int m  = id / HID;
            int j  = id - m * HID;
            float xv = xtile[m * T_LEN + t];
            const float* gr = gates + m * GPAD + j;
            float gi = gr[0]   + bias1[j]       + xv * wih1[j];
            float gf = gr[50]  + bias1[50 + j]  + xv * wih1[50 + j];
            float gg = gr[100] + bias1[100 + j] + xv * wih1[100 + j];
            float go = gr[150] + bias1[150 + j] + xv * wih1[150 + j];
            float ig = fsig(gi), fg = fsig(gf), gn = ftanh(gg), og = fsig(go);
            float c = fg * c1[q] + ig * gn;
            c1[q] = c;
            float h = og * ftanh(c);
            hbuf1[h_addr(m, j)] = (_Float16)h;   // A-fragment layout for layer 2
        }

        // ======= Layer 2: gates2 = h1_new @ W_ih2^T + h2 @ W_hh2^T =======
        v16h p0 = *(const v16h*)(hbuf1 + lane * 16);
        v16h p1 = *(const v16h*)(hbuf1 + (32 + lane) * 16);
        v16h q0 = *(const v16h*)(hbuf2 + lane * 16);
        v16h q1 = *(const v16h*)(hbuf2 + (32 + lane) * 16);
#pragma unroll
        for (int tl = 0; tl < NTILE; ++tl) {
            v16h bi0 = *(const v16h*)(wI2 + ((tl * 2 + 0) * 32 + lane) * 16);
            v16h bi1 = *(const v16h*)(wI2 + ((tl * 2 + 1) * 32 + lane) * 16);
            v16h bh0 = *(const v16h*)(wH2 + ((tl * 2 + 0) * 32 + lane) * 16);
            v16h bh1 = *(const v16h*)(wH2 + ((tl * 2 + 1) * 32 + lane) * 16);
            v8f acc = {0.f, 0.f, 0.f, 0.f, 0.f, 0.f, 0.f, 0.f};
            acc = __builtin_amdgcn_wmma_f32_16x16x32_f16(false, p0, false, bi0,
                                                         (short)0, acc, false, false);
            acc = __builtin_amdgcn_wmma_f32_16x16x32_f16(false, p1, false, bi1,
                                                         (short)0, acc, false, false);
            acc = __builtin_amdgcn_wmma_f32_16x16x32_f16(false, q0, false, bh0,
                                                         (short)0, acc, false, false);
            acc = __builtin_amdgcn_wmma_f32_16x16x32_f16(false, q1, false, bh1,
                                                         (short)0, acc, false, false);
            int col = tl * 16 + colL;
#pragma unroll
            for (int v = 0; v < 8; ++v) gates[(row0 + v) * GPAD + col] = acc[v];
        }
        // ---- layer-2 elementwise ----
#pragma unroll
        for (int q = 0; q < 25; ++q) {
            int id = lane + 32 * q;
            int m  = id / HID;
            int j  = id - m * HID;
            float* gr = gates + m * GPAD + j;
            float gi = gr[0]   + bias2[j];
            float gf = gr[50]  + bias2[50 + j];
            float gg = gr[100] + bias2[100 + j];
            float go = gr[150] + bias2[150 + j];
            float ig = fsig(gi), fg = fsig(gf), gn = ftanh(gg), og = fsig(go);
            float c = fg * c2[q] + ig * gn;
            c2[q] = c;
            float h = og * ftanh(c);
            hbuf2[h_addr(m, j)] = (_Float16)h;   // recurrent input for next step
            gr[0] = h;                            // keep f32 h2 for the FC readout
        }
    }

    // ---- final FC: out[m] = h2_last . fc_W + fc_b ----
    if (lane < MROWS) {
        float s = fcb[0];
#pragma unroll 10
        for (int j = 0; j < HID; ++j) s += gates[lane * GPAD + j] * fcw[j];
        out[bm + lane] = s;
    }
}

extern "C" void kernel_launch(void* const* d_in, const int* in_sizes, int n_in,
                              void* d_out, int out_size, void* d_ws, size_t ws_size,
                              hipStream_t stream) {
    const float* x    = (const float*)d_in[0];
    const float* Wih1 = (const float*)d_in[1];
    const float* Whh1 = (const float*)d_in[2];
    const float* bih1 = (const float*)d_in[3];
    const float* bhh1 = (const float*)d_in[4];
    const float* Wih2 = (const float*)d_in[5];
    const float* Whh2 = (const float*)d_in[6];
    const float* bih2 = (const float*)d_in[7];
    const float* bhh2 = (const float*)d_in[8];
    const float* fcW  = (const float*)d_in[9];
    const float* fcb  = (const float*)d_in[10];
    float* out = (float*)d_out;

    // 2048 batch rows / 16 per wave = 128 single-wave workgroups.
    lstm_fused<<<dim3(128), dim3(32), SMEM_BYTES, stream>>>(
        x, Wih1, Whh1, bih1, bhh1, Wih2, Whh2, bih2, bhh2, fcW, fcb, out);
}
